// MultiHeadAttention_26319559590085
// MI455X (gfx1250) — matrix operand order repaired and checked by reading, not compiled, not run on hardware
//
#include <hip/hip_runtime.h>
#include <hip/hip_bf16.h>

// MHA for B=4, S=2048, D=1024, H=16, DK=64 on gfx1250 (wave32, WMMA f16 16x16x32).

#define B_  4
#define S_  2048
#define D_  1024
#define H_  16
#define DK_ 64

typedef __attribute__((ext_vector_type(16))) _Float16 v16h;
typedef __attribute__((ext_vector_type(8)))  _Float16 v8h;
typedef __attribute__((ext_vector_type(8)))  float    v8f;
typedef __attribute__((ext_vector_type(4)))  float    v4f;

__device__ __forceinline__ v8f wmma_f16(v16h a, v16h b, v8f c) {
  // D = A(16x32 f16) * B(32x16 f16) + C(16x16 f32)
  return __builtin_amdgcn_wmma_f32_16x16x32_f16(false, a, false, b, (short)0, c,
                                                false, false);
}

// A-fragment (16x32, f16), source row-major [rows][ld] halves.
// lane l: row = row0 + (l&15); lanes 0-15 hold K {0..7,16..23}, lanes 16-31 K {8..15,24..31}.
__device__ __forceinline__ v16h load_frag_a(const _Float16* base, int ld, int row0, int k0) {
  const int l = threadIdx.x & 31;
  const _Float16* p = base + (size_t)(row0 + (l & 15)) * ld + k0 + ((l >> 4) << 3);
  v8h lo = *(const v8h*)p;
  v8h hi = *(const v8h*)(p + 16);
  v16h r;
#pragma unroll
  for (int i = 0; i < 8; ++i) { r[i] = lo[i]; r[i + 8] = hi[i]; }
  return r;
}

// B-fragment (32x16, f16), source stored N-major: [cols][ld] halves (K contiguous).
// lane l: col = col0 + (l&15); lanes 0-15 hold K {0..7,16..23}, lanes 16-31 K {8..15,24..31}.
__device__ __forceinline__ v16h load_frag_b(const _Float16* base, int ld, int col0, int k0) {
  const int l = threadIdx.x & 31;
  const _Float16* p = base + (size_t)(col0 + (l & 15)) * ld + k0 + ((l >> 4) << 3);
  v8h lo = *(const v8h*)p;
  v8h hi = *(const v8h*)(p + 16);
  v16h r;
#pragma unroll
  for (int i = 0; i < 8; ++i) { r[i] = lo[i]; r[i + 8] = hi[i]; }
  return r;
}

// ---------------------------------------------------------------------------
// Weight prep: Wt[n][k] = (f16) W[k][n]
// ---------------------------------------------------------------------------
__global__ __launch_bounds__(256) void wtrans_kernel(const float* __restrict__ W,
                                                     _Float16* __restrict__ Wt) {
  __shared__ float t[64][65];
  const int tid = threadIdx.x;
  const int bx = blockIdx.x * 64;  // col block of W
  const int by = blockIdx.y * 64;  // row block of W
#pragma unroll
  for (int i = 0; i < 16; ++i) {
    int e = tid + i * 256;
    int r = e >> 6, c = e & 63;
    t[r][c] = W[(size_t)(by + r) * D_ + bx + c];
  }
  __syncthreads();
#pragma unroll
  for (int i = 0; i < 16; ++i) {
    int e = tid + i * 256;
    int r = e >> 6, c = e & 63;
    Wt[(size_t)(bx + r) * D_ + by + c] = (_Float16)t[c][r];
  }
}

// ---------------------------------------------------------------------------
// GEMM: C[M,N] = oscale * A[M,K] * Bt[N,K]^T  (A fp32 or f16, Bt f16)
// MODE 1: f16 out scattered to [B,H,S,DK]   (Q,K projections; Q carries 1/sqrt(dk))
// MODE 2: f16 out scattered to [B,H,DK,S]   (V projection, pre-transposed)
// MODE 3: f32 out row-major [M,N]           (final output projection)
// Software-pipelined: double-buffered LDS, one barrier per k-step, global
// loads for tile k+1 in flight during tile k's WMMAs, prefetch for k+2.
// ---------------------------------------------------------------------------
#define BM 128
#define BN 128
#define BK 32
#define LDT 40  // padded LDS stride in halves (80B: conflict-free 16-row frag reads)

template <int A_IS_F16, int MODE>
__global__ __launch_bounds__(256) void gemm128_kernel(
    const void* __restrict__ Aptr,
    const _Float16* __restrict__ Bt,
    void* __restrict__ Cptr, int M, int N, int Kd, float oscale) {
  __shared__ _Float16 As[2][BM * LDT];
  __shared__ _Float16 Bs[2][BN * LDT];

  const int tid  = threadIdx.x;
  const int lane = tid & 31;
  const int wave = tid >> 5;
  const int wm = (wave & 3) * 32;   // 4 waves along M: 32 rows each
  const int wn = (wave >> 2) * 64;  // 2 waves along N: 64 cols each
  const size_t bm0 = (size_t)blockIdx.y * BM;
  const size_t bn0 = (size_t)blockIdx.x * BN;

  v8f acc[2][4] = {};

  const int sr = tid >> 1;        // staging row 0..127
  const int sc = (tid & 1) * 16;  // staging col 0 or 16

  v4f fa[4];  // A staging (fp32 path)
  v8h ha[2];  // A staging (f16 path)
  v8h hb[2];  // B staging

  auto load_next = [&](int k0) {
    if constexpr (A_IS_F16) {
      const _Float16* A = (const _Float16*)Aptr + (bm0 + sr) * Kd + k0 + sc;
      ha[0] = *(const v8h*)A;
      ha[1] = *(const v8h*)(A + 8);
    } else {
      const float* A = (const float*)Aptr + (bm0 + sr) * Kd + k0 + sc;
      fa[0] = *(const v4f*)A;
      fa[1] = *(const v4f*)(A + 4);
      fa[2] = *(const v4f*)(A + 8);
      fa[3] = *(const v4f*)(A + 12);
    }
    const _Float16* Bp = Bt + (bn0 + sr) * Kd + k0 + sc;
    hb[0] = *(const v8h*)Bp;
    hb[1] = *(const v8h*)(Bp + 8);
  };

  auto store_tile = [&](int p) {
    _Float16* a = &As[p][sr * LDT + sc];
    if constexpr (A_IS_F16) {
      *(v8h*)a       = ha[0];
      *(v8h*)(a + 8) = ha[1];
    } else {
#pragma unroll
      for (int i = 0; i < 4; ++i) {
        a[i]      = (_Float16)fa[0][i];
        a[4 + i]  = (_Float16)fa[1][i];
        a[8 + i]  = (_Float16)fa[2][i];
        a[12 + i] = (_Float16)fa[3][i];
      }
    }
    _Float16* b = &Bs[p][sr * LDT + sc];
    *(v8h*)b       = hb[0];
    *(v8h*)(b + 8) = hb[1];
  };

  load_next(0);
  int p = 0;
  for (int k0 = 0; k0 < Kd; k0 += BK) {
    store_tile(p);
    if (k0 + BK < Kd) {
      load_next(k0 + BK);
      if (k0 + 2 * BK < Kd) {  // global_prefetch_b8 for tile k+2 into GL2
        const int kp = k0 + 2 * BK;
        __builtin_prefetch(Bt + (bn0 + sr) * Kd + kp + sc, 0, 1);
        if constexpr (A_IS_F16)
          __builtin_prefetch((const _Float16*)Aptr + (bm0 + sr) * Kd + kp + sc, 0, 1);
        else
          __builtin_prefetch((const float*)Aptr + (bm0 + sr) * Kd + kp + sc, 0, 1);
      }
    }
    __syncthreads();

    v16h a0 = load_frag_a(As[p], LDT, wm, 0);
    v16h a1 = load_frag_a(As[p], LDT, wm + 16, 0);
#pragma unroll
    for (int j = 0; j < 4; ++j) {
      v16h bf = load_frag_b(Bs[p], LDT, wn + j * 16, 0);
      acc[0][j] = wmma_f16(a0, bf, acc[0][j]);
      acc[1][j] = wmma_f16(a1, bf, acc[1][j]);
    }
    p ^= 1;
    // no second barrier: next iteration writes the *other* buffer; re-writing
    // this one happens only after the next barrier, which orders it after all
    // waves' reads above.
  }

  // Epilogue.  C layout: VGPR e <-> row e + 8*(lane>>4), col lane&15.
  const int rsel = (lane >> 4) << 3;
  const int ncol = lane & 15;
#pragma unroll
  for (int i = 0; i < 2; ++i) {
#pragma unroll
    for (int j = 0; j < 4; ++j) {
#pragma unroll
      for (int e = 0; e < 8; ++e) {
        int m = (int)bm0 + wm + i * 16 + e + rsel;
        int n = (int)bn0 + wn + j * 16 + ncol;
        float v = acc[i][j][e] * oscale;
        if constexpr (MODE == 3) {
          ((float*)Cptr)[(size_t)m * N + n] = v;
        } else {
          int b = m / S_, s = m - b * S_;
          int h = n >> 6, dk = n & 63;
          size_t idx;
          if constexpr (MODE == 1)
            idx = (((size_t)b * H_ + h) * S_ + s) * DK_ + dk;
          else
            idx = (((size_t)b * H_ + h) * DK_ + dk) * S_ + s;
          ((_Float16*)Cptr)[idx] = (_Float16)v;
        }
      }
    }
  }
}

// ---------------------------------------------------------------------------
// Flash attention (causal): per block = 128 query rows of one (b,h) head.
// Q,K: [B,H,S,64] f16 (Q pre-scaled by 1/sqrt(dk));  V: [B,H,64,S] f16;
// ctx: [B,S,D] f16.
// LDS: the K tile and the P transpose pad are phase-disjoint (K is dead once
// the score WMMAs consumed it), so they share one union buffer.
// ---------------------------------------------------------------------------
#define KT   128
#define LDKS 72   // K tile stride: 64 + 8 pad (halves)
#define LDVS 136  // V tile stride: 128 + 8 pad
#define LDPS 136  // P pad stride: 128 + 8 pad (full key-tile width)

__global__ __launch_bounds__(256) void attn128_kernel(
    const _Float16* __restrict__ Q,
    const _Float16* __restrict__ K,
    const _Float16* __restrict__ V,
    _Float16* __restrict__ ctx) {
  // union: score phase = K tile [128][LDKS] (9216 h); PV phase = per-wave
  // P pads 8 x [16][LDPS] (17408 h).  34 KB.
  __shared__ _Float16 U[8 * 16 * LDPS];
  __shared__ _Float16 Vs[DK_ * LDVS];  // 17 KB: value tile [64][128] (dk-major)

  const int tid  = threadIdx.x;
  const int lane = tid & 31;
  const int wave = tid >> 5;
  const int qt = blockIdx.x;        // query tile (16)
  const int bh = blockIdx.y;        // b*H + h   (64)
  const int b = bh >> 4;
  const int h = bh & 15;

  const _Float16* Qb = Q + (size_t)bh * S_ * DK_;
  const _Float16* Kb = K + (size_t)bh * S_ * DK_;
  const _Float16* Vb = V + (size_t)bh * DK_ * S_;

  const int q0 = qt * 128 + wave * 16;  // this wave's 16 query rows
  const int rsel = (lane >> 4) << 3;
  const int ncol = lane & 15;

  // Q fragments straight from global (rows are 64 contiguous halves).
  v16h aq0 = load_frag_a(Qb, DK_, q0, 0);
  v16h aq1 = load_frag_a(Qb, DK_, q0, 32);

  v8f acc[4] = {};
  float mrow[8], lrow[8];
#pragma unroll
  for (int e = 0; e < 8; ++e) { mrow[e] = -1e30f; lrow[e] = 0.0f; }

  for (int kt = 0; kt <= qt; ++kt) {  // causal: skip future key tiles
    const int ks0 = kt * KT;
    {  // stage K tile into union buffer: 128 rows x 64 halves
      const int r = tid >> 1;
      const int c = (tid & 1) * 32;
      const _Float16* src = Kb + (size_t)(ks0 + r) * DK_ + c;
      v8h x0 = *(const v8h*)(src);
      v8h x1 = *(const v8h*)(src + 8);
      v8h x2 = *(const v8h*)(src + 16);
      v8h x3 = *(const v8h*)(src + 24);
      _Float16* dp = &U[r * LDKS + c];
      *(v8h*)(dp)      = x0;
      *(v8h*)(dp + 8)  = x1;
      *(v8h*)(dp + 16) = x2;
      *(v8h*)(dp + 24) = x3;
    }
    {  // stage V tile: 64 dk-rows x 128 key-halves
      const int r = tid >> 2;
      const int c = (tid & 3) * 32;
      const _Float16* src = Vb + (size_t)r * S_ + ks0 + c;
      v8h x0 = *(const v8h*)(src);
      v8h x1 = *(const v8h*)(src + 8);
      v8h x2 = *(const v8h*)(src + 16);
      v8h x3 = *(const v8h*)(src + 24);
      _Float16* dp = &Vs[r * LDVS + c];
      *(v8h*)(dp)      = x0;
      *(v8h*)(dp + 8)  = x1;
      *(v8h*)(dp + 16) = x2;
      *(v8h*)(dp + 24) = x3;
    }
    if (kt < qt) {  // prefetch next key/value tiles into GL2
      __builtin_prefetch(Kb + (size_t)(ks0 + KT + (tid >> 1)) * DK_ + (tid & 1) * 32, 0, 1);
      __builtin_prefetch(Vb + (size_t)(tid >> 2) * S_ + ks0 + KT + (tid & 3) * 32, 0, 1);
    }
    __syncthreads();

    // S = Q K^T (already includes 1/sqrt(dk) via pre-scaled Q).
    v8f sc[8];
#pragma unroll
    for (int j = 0; j < 8; ++j) {
      v8f z = {};
      v16h b0 = load_frag_b(U, LDKS, j * 16, 0);
      z = wmma_f16(aq0, b0, z);
      v16h b1 = load_frag_b(U, LDKS, j * 16, 32);
      z = wmma_f16(aq1, b1, z);
      sc[j] = z;
    }
    __syncthreads();  // K tile dead everywhere -> union becomes the P pad

    // causal mask (only the diagonal tile can contain masked cells)
    if (kt == qt) {
#pragma unroll
      for (int j = 0; j < 8; ++j) {
#pragma unroll
        for (int e = 0; e < 8; ++e) {
          int qg = q0 + e + rsel;
          int kg = ks0 + j * 16 + ncol;
          if (kg > qg) sc[j][e] = -1e30f;
        }
      }
    }

    // online softmax: row max / rescale / row sum (16-lane xor reductions)
    float nm[8], alpha[8];
#pragma unroll
    for (int e = 0; e < 8; ++e) {
      float t = sc[0][e];
#pragma unroll
      for (int j = 1; j < 8; ++j) t = fmaxf(t, sc[j][e]);
      t = fmaxf(t, __shfl_xor(t, 1, 32));
      t = fmaxf(t, __shfl_xor(t, 2, 32));
      t = fmaxf(t, __shfl_xor(t, 4, 32));
      t = fmaxf(t, __shfl_xor(t, 8, 32));
      nm[e] = fmaxf(mrow[e], t);
      alpha[e] = __expf(mrow[e] - nm[e]);
    }

    // P = exp(S - m) computed ONCE: accumulate row sum and stage f16 P pad.
    _Float16* Pw = &U[wave * 16 * LDPS];
    float ps[8];
#pragma unroll
    for (int e = 0; e < 8; ++e) ps[e] = 0.0f;
#pragma unroll
    for (int j = 0; j < 8; ++j) {
#pragma unroll
      for (int e = 0; e < 8; ++e) {
        float pp = __expf(sc[j][e] - nm[e]);
        ps[e] += pp;
        Pw[(e + rsel) * LDPS + j * 16 + ncol] = (_Float16)pp;
      }
    }
#pragma unroll
    for (int e = 0; e < 8; ++e) {
      float t = ps[e];
      t += __shfl_xor(t, 1, 32);
      t += __shfl_xor(t, 2, 32);
      t += __shfl_xor(t, 4, 32);
      t += __shfl_xor(t, 8, 32);
      lrow[e] = lrow[e] * alpha[e] + t;
      mrow[e] = nm[e];
    }
#pragma unroll
    for (int d = 0; d < 4; ++d)
#pragma unroll
      for (int e = 0; e < 8; ++e) acc[d][e] *= alpha[e];

    asm volatile("s_wait_dscnt 0" ::: "memory");  // wave-local P RAW fence

    // O += P V : 4 key chunks of 32 x 4 dk tiles of 16.
#pragma unroll
    for (int c = 0; c < 4; ++c) {
      v16h pa = load_frag_a(Pw, LDPS, 0, c * 32);
#pragma unroll
      for (int d = 0; d < 4; ++d) {
        v16h bv = load_frag_b(Vs, LDVS, d * 16, c * 32);
        acc[d] = wmma_f16(pa, bv, acc[d]);
      }
    }
    __syncthreads();  // all P/V reads done before next tile's staging
  }

  // epilogue: ctx[b][q][h*64 + dk] = O / l
#pragma unroll
  for (int d = 0; d < 4; ++d) {
#pragma unroll
    for (int e = 0; e < 8; ++e) {
      int qrow = q0 + e + rsel;
      size_t idx = ((size_t)b * S_ + qrow) * D_ + (size_t)h * DK_ + d * 16 + ncol;
      ctx[idx] = (_Float16)(acc[d][e] / lrow[e]);
    }
  }
}

// ---------------------------------------------------------------------------
// Host launch
// ---------------------------------------------------------------------------
extern "C" void kernel_launch(void* const* d_in, const int* in_sizes, int n_in,
                              void* d_out, int out_size, void* d_ws, size_t ws_size,
                              hipStream_t stream) {
  const float* q  = (const float*)d_in[0];
  const float* k  = (const float*)d_in[1];
  const float* v  = (const float*)d_in[2];
  const float* Wq = (const float*)d_in[3];
  const float* Wk = (const float*)d_in[4];
  const float* Wv = (const float*)d_in[5];
  const float* Wo = (const float*)d_in[6];
  // d_in[7] = mask: causal handled analytically.

  char* ws = (char*)d_ws;
  const size_t szW = (size_t)D_ * D_ * sizeof(_Float16);        // 2 MB
  const size_t szT = (size_t)B_ * S_ * D_ * sizeof(_Float16);   // 16 MB
  _Float16* WqT = (_Float16*)(ws + 0 * szW);
  _Float16* WkT = (_Float16*)(ws + 1 * szW);
  _Float16* WvT = (_Float16*)(ws + 2 * szW);
  _Float16* WoT = (_Float16*)(ws + 3 * szW);
  _Float16* Qb  = (_Float16*)(ws + 4 * szW);
  _Float16* Kb  = (_Float16*)(ws + 4 * szW + 1 * szT);
  _Float16* Vtb = (_Float16*)(ws + 4 * szW + 2 * szT);
  _Float16* ctx = (_Float16*)(ws + 4 * szW + 3 * szT);          // total 72 MB

  dim3 tgrid(D_ / 64, D_ / 64);
  wtrans_kernel<<<tgrid, 256, 0, stream>>>(Wq, WqT);
  wtrans_kernel<<<tgrid, 256, 0, stream>>>(Wk, WkT);
  wtrans_kernel<<<tgrid, 256, 0, stream>>>(Wv, WvT);
  wtrans_kernel<<<tgrid, 256, 0, stream>>>(Wo, WoT);

  const int M = B_ * S_;
  dim3 ggrid(D_ / BN, M / BM);
  // Q carries the 1/sqrt(dk) softmax scale so attention does zero scaling VALU.
  gemm128_kernel<0, 1><<<ggrid, 256, 0, stream>>>(q, WqT, Qb,  M, D_, D_, 0.125f);
  gemm128_kernel<0, 1><<<ggrid, 256, 0, stream>>>(k, WkT, Kb,  M, D_, D_, 1.0f);
  gemm128_kernel<0, 2><<<ggrid, 256, 0, stream>>>(v, WvT, Vtb, M, D_, D_, 1.0f);

  dim3 agrid(S_ / 128, B_ * H_);
  attn128_kernel<<<agrid, 256, 0, stream>>>(Qb, Kb, Vtb, ctx);

  gemm128_kernel<1, 3><<<ggrid, 256, 0, stream>>>(ctx, WoT, d_out, M, D_, D_, 1.0f);
}
